// DIRU_2843268350850
// MI455X (gfx1250) — matrix operand order repaired and checked by reading, not compiled, not run on hardware
//
#include <hip/hip_runtime.h>
#include <hip/hip_bf16.h>
#include <cstddef>

typedef __attribute__((ext_vector_type(16))) _Float16 v16h;
typedef __attribute__((ext_vector_type(8)))  _Float16 v8h;
typedef __attribute__((ext_vector_type(4)))  _Float16 v4h;
typedef __attribute__((ext_vector_type(8)))  float    v8f;

// Explicit global-address-space pointer types: force global_load_b128 (pure
// LOADcnt) instead of flat_load_b128 (LOADcnt+DScnt, aperture checks).
typedef __attribute__((address_space(1))) const _Float16 ghalf_t;
typedef __attribute__((address_space(1))) const v8h      gv8h_t;

// Problem constants
#define B_    1024
#define T_    512
#define I_    40
#define H_    256
#define K_    4
#define KH_   1024      // K*H
#define O_    16
#define KA_   320       // concat K dim: 256 (h) + 64 (x padded from 40)
#define KCHUNKS 10      // 320 / 32
#define TILE_M 16       // batch rows per block
#define WCAT_BYTES (KH_ * KA_ * 2)

// LDS strides (padded to avoid bank conflicts)
#define A_STRIDE    328   // halves (16B-aligned rows, conflict-free b128 A loads)
#define OUTS_STRIDE 1028  // floats
#define H32_STRIDE  260   // floats

__device__ __forceinline__ float fast_tanh(float x) {
    float e = __expf(2.0f * x);          // stable at +/-inf
    return 1.0f - 2.0f / (e + 1.0f);
}

// Build combined f16 weight Wcat[1024][320] = [W_rec (256) | W_in_flat (40) | zeros (24)]
__global__ void build_wcat_kernel(const float* __restrict__ W_rec,
                                  const float* __restrict__ W_in,
                                  _Float16* __restrict__ wcat) {
    int idx = blockIdx.x * blockDim.x + threadIdx.x;
    if (idx >= KH_ * KA_) return;
    int j = idx / KA_;
    int c = idx - j * KA_;
    float v = 0.0f;
    if (c < H_)            v = W_rec[j * H_ + c];
    else if (c < H_ + I_)  v = W_in[j * I_ + (c - H_)];
    wcat[idx] = (_Float16)v;
}

// Per-timestep weight-base pointer table (all entries identical). This makes the
// B-fragment addresses depend on a load inside iteration t, so LLVM cannot hoist
// the 80 weight fragments out of the t-loop (which would spill to scratch).
__global__ void fill_tab_kernel(const _Float16* wcat, const _Float16** tab) {
    int i = blockIdx.x * blockDim.x + threadIdx.x;
    if (i < T_) tab[i] = wcat;
}

__global__ __launch_bounds__(256) void diru_persistent_kernel(
    const float* __restrict__ x,       // [B, T, I]
    const float* __restrict__ b_in,    // [K*H] flat
    const float* __restrict__ b_rec,   // [K*H]
    const float* __restrict__ W_gate,  // [K, K*H]
    const float* __restrict__ b_gate,  // [K]
    const float* __restrict__ W_fc,    // [O, H]
    const float* __restrict__ b_fc,    // [O]
    const _Float16* const* __restrict__ wtab, // [T] -> Wcat base
    float* __restrict__ out)           // [B, O]
{
    __shared__ _Float16 A_lds[TILE_M * A_STRIDE];        // [h | x_t] tile, f16
    __shared__ float    outs_lds[TILE_M * OUTS_STRIDE];  // tanh outputs [16][1024]
    __shared__ float    Wg_lds[K_ * KH_];                // gate weights
    __shared__ float    biasc_lds[KH_];                  // b_rec + b_in
    __shared__ float    hf32_lds[TILE_M * H32_STRIDE];   // f32 hidden state
    __shared__ float    logits_lds[TILE_M * K_];
    __shared__ float    w_lds[TILE_M * K_];

    const int tid   = threadIdx.x;
    const int lane  = tid & 31;
    const int wave  = tid >> 5;
    const int l16   = lane & 15;
    const bool hiHalf = (lane >= 16);
    const int bbase = blockIdx.x * TILE_M;
    const int nwave = wave * 128;   // this wave's 128 output columns

    // ---- preamble ----
    for (int i = tid; i < TILE_M * A_STRIDE; i += 256) A_lds[i] = (_Float16)0.0f;
    for (int i = tid; i < K_ * KH_; i += 256) Wg_lds[i] = W_gate[i];
    for (int i = tid; i < KH_; i += 256) biasc_lds[i] = b_rec[i] + b_in[i];
    for (int i = tid; i < TILE_M * I_; i += 256) {
        int row = i / I_, ii = i - row * I_;
        A_lds[row * A_STRIDE + H_ + ii] =
            (_Float16)x[((size_t)(bbase + row) * T_ + 0) * I_ + ii];
    }
    const float bgk = b_gate[(tid >> 2) & 3];
    __syncthreads();

    const int aoff = hiHalf ? 8 : 0;     // A-frag K sub-offset per half-wave
    const int boffH = hiHalf ? 8 : 0;    // B-frag K sub-offset per half-wave

    // Loop-invariant per-lane state hoisted out of the t-loop:
    float bv[8];                         // bias for acc init
    int   subOff[8];                     // B-frag per-lane offsets (halves)
    #pragma unroll
    for (int sub = 0; sub < 8; ++sub) {
        bv[sub] = biasc_lds[nwave + sub * 16 + l16];
        subOff[sub] = (nwave + sub * 16 + l16) * KA_ + boffH;
    }

    for (int t = 0; t < T_; ++t) {
        ghalf_t* wbase = (ghalf_t*)wtab[t];   // per-t load defeats LICM; AS1 cast
                                              // => global_load, pure LOADcnt

        // ================= fused GEMM: [16 x 320] x [320 x 1024] ==============
        v8f acc[8];
        #pragma unroll
        for (int sub = 0; sub < 8; ++sub)
            acc[sub] = (v8f){bv[sub], bv[sub], bv[sub], bv[sub],
                             bv[sub], bv[sub], bv[sub], bv[sub]};

        v16h bbuf[2][4];   // double-buffered B fragments (groups of 4 subtiles)
        v16h abuf[2];      // double-buffered A fragment

        // prologue: A chunk 0, B group 0
        {
            const _Float16* ap = &A_lds[l16 * A_STRIDE + aoff];
            v8h alo = *(const v8h*)ap;
            v8h ahi = *(const v8h*)(ap + 16);
            abuf[0] = __builtin_shufflevector(alo, ahi,
                0,1,2,3,4,5,6,7,8,9,10,11,12,13,14,15);
            #pragma unroll
            for (int j = 0; j < 4; ++j) {
                ghalf_t* bp = wbase + subOff[j];
                v8h blo = *(gv8h_t*)bp;
                v8h bhi = *(gv8h_t*)(bp + 16);
                bbuf[0][j] = __builtin_shufflevector(blo, bhi,
                    0,1,2,3,4,5,6,7,8,9,10,11,12,13,14,15);
            }
        }

        // 20 groups = 10 K-chunks x 2 half-chunks of 4 subtiles
        #pragma unroll
        for (int g = 0; g < 20; ++g) {
            const int kc  = g >> 1;
            const int s0  = (g & 1) * 4;
            // prefetch next B group
            if (g < 19) {
                const int nkc = (g + 1) >> 1;
                const int ns0 = ((g + 1) & 1) * 4;
                #pragma unroll
                for (int j = 0; j < 4; ++j) {
                    ghalf_t* bp = wbase + subOff[ns0 + j] + nkc * 32;
                    v8h blo = *(gv8h_t*)bp;
                    v8h bhi = *(gv8h_t*)(bp + 16);
                    bbuf[(g + 1) & 1][j] = __builtin_shufflevector(blo, bhi,
                        0,1,2,3,4,5,6,7,8,9,10,11,12,13,14,15);
                }
            }
            // prefetch next A chunk (during the second half-chunk)
            if ((g & 1) == 1 && kc < 9) {
                const _Float16* ap = &A_lds[l16 * A_STRIDE + (kc + 1) * 32 + aoff];
                v8h alo = *(const v8h*)ap;
                v8h ahi = *(const v8h*)(ap + 16);
                abuf[(kc + 1) & 1] = __builtin_shufflevector(alo, ahi,
                    0,1,2,3,4,5,6,7,8,9,10,11,12,13,14,15);
            }
            #pragma unroll
            for (int j = 0; j < 4; ++j) {
                acc[s0 + j] = __builtin_amdgcn_wmma_f32_16x16x32_f16(
                    false, abuf[kc & 1], false, bbuf[g & 1][j],
                    (short)0, acc[s0 + j], false, false);
            }
        }

        // ================= tanh + scatter to LDS (C/D layout) =================
        #pragma unroll
        for (int sub = 0; sub < 8; ++sub) {
            const int nb = nwave + sub * 16 + l16;
            #pragma unroll
            for (int r = 0; r < 8; ++r) {
                int m = r + (hiHalf ? 8 : 0);
                outs_lds[m * OUTS_STRIDE + nb] = fast_tanh(acc[sub][r]);
            }
        }
        __syncthreads();

        // ========= gate logits: 64 (row,k) pairs, 4 threads x 256 contiguous ==
        {
            const int pair = tid >> 2, subl = tid & 3;
            const int row = pair >> 2, k = pair & 3;
            const float4* o4 = (const float4*)(&outs_lds[row * OUTS_STRIDE] + subl * 256);
            const float4* w4 = (const float4*)(&Wg_lds[k * KH_] + subl * 256);
            float s = 0.0f;
            #pragma unroll 8
            for (int i = 0; i < 64; ++i) {
                float4 o = o4[i], w = w4[i];
                s += o.x * w.x + o.y * w.y + o.z * w.z + o.w * w.w;
            }
            s += __shfl_xor(s, 1);
            s += __shfl_xor(s, 2);
            if (subl == 0) logits_lds[pair] = s + bgk;
        }
        __syncthreads();

        // ================= softmax over K=4 (threads 0..15) ===================
        if (tid < TILE_M) {
            float l0 = logits_lds[tid * 4 + 0];
            float l1 = logits_lds[tid * 4 + 1];
            float l2 = logits_lds[tid * 4 + 2];
            float l3 = logits_lds[tid * 4 + 3];
            float mx = fmaxf(fmaxf(l0, l1), fmaxf(l2, l3));
            float e0 = __expf(l0 - mx), e1 = __expf(l1 - mx);
            float e2 = __expf(l2 - mx), e3 = __expf(l3 - mx);
            float inv = 1.0f / (e0 + e1 + e2 + e3);
            w_lds[tid * 4 + 0] = e0 * inv;
            w_lds[tid * 4 + 1] = e1 * inv;
            w_lds[tid * 4 + 2] = e2 * inv;
            w_lds[tid * 4 + 3] = e3 * inv;
        }
        __syncthreads();

        // ================= h_new = sum_k w_k * outs_k; stage x_{t+1} ==========
        {
            const int row = tid >> 4;
            const int cseg = (tid & 15) * 16;
            const float w0 = w_lds[row * 4 + 0];
            const float w1 = w_lds[row * 4 + 1];
            const float w2 = w_lds[row * 4 + 2];
            const float w3 = w_lds[row * 4 + 3];
            const float* orow = &outs_lds[row * OUTS_STRIDE];
            #pragma unroll
            for (int q = 0; q < 4; ++q) {
                const int c = cseg + q * 4;
                float4 o0 = *(const float4*)(orow + c);
                float4 o1 = *(const float4*)(orow + H_ + c);
                float4 o2 = *(const float4*)(orow + 2 * H_ + c);
                float4 o3 = *(const float4*)(orow + 3 * H_ + c);
                float4 hv;
                hv.x = w0 * o0.x + w1 * o1.x + w2 * o2.x + w3 * o3.x;
                hv.y = w0 * o0.y + w1 * o1.y + w2 * o2.y + w3 * o3.y;
                hv.z = w0 * o0.z + w1 * o1.z + w2 * o2.z + w3 * o3.z;
                hv.w = w0 * o0.w + w1 * o1.w + w2 * o2.w + w3 * o3.w;
                *(float4*)(&hf32_lds[row * H32_STRIDE + c]) = hv;
                v4h hh = {(_Float16)hv.x, (_Float16)hv.y,
                          (_Float16)hv.z, (_Float16)hv.w};
                *(v4h*)(&A_lds[row * A_STRIDE + c]) = hh;
            }
            if (t + 1 < T_) {
                for (int i = tid; i < TILE_M * I_; i += 256) {
                    int row2 = i / I_, ii = i - row2 * I_;
                    A_lds[row2 * A_STRIDE + H_ + ii] =
                        (_Float16)x[((size_t)(bbase + row2) * T_ + (t + 1)) * I_ + ii];
                }
            }
        }
        __syncthreads();
    }

    // ================= final FC: out = h @ W_fc.T + b_fc ======================
    {
        const int row = tid >> 4, o = tid & 15;
        const float* hr = &hf32_lds[row * H32_STRIDE];
        const float* wf = &W_fc[o * H_];
        float s = b_fc[o];
        #pragma unroll 8
        for (int c = 0; c < H_; ++c) s += hr[c] * wf[c];
        out[(size_t)(bbase + row) * O_ + o] = s;
    }
}

extern "C" void kernel_launch(void* const* d_in, const int* in_sizes, int n_in,
                              void* d_out, int out_size, void* d_ws, size_t ws_size,
                              hipStream_t stream) {
    (void)in_sizes; (void)n_in; (void)out_size; (void)ws_size;
    const float* x      = (const float*)d_in[0];
    const float* W_in   = (const float*)d_in[1];
    const float* b_in   = (const float*)d_in[2];
    const float* W_rec  = (const float*)d_in[3];
    const float* b_rec  = (const float*)d_in[4];
    const float* W_gate = (const float*)d_in[5];
    const float* b_gate = (const float*)d_in[6];
    const float* W_fc   = (const float*)d_in[7];
    const float* b_fc   = (const float*)d_in[8];

    _Float16* wcat = (_Float16*)d_ws;                               // 640 KB
    const _Float16** wtab = (const _Float16**)((char*)d_ws + WCAT_BYTES); // 4 KB

    {
        int total = KH_ * KA_;
        int blocks = (total + 255) / 256;
        build_wcat_kernel<<<blocks, 256, 0, stream>>>(W_rec, W_in, wcat);
    }
    fill_tab_kernel<<<2, 256, 0, stream>>>(wcat, wtab);
    diru_persistent_kernel<<<B_ / TILE_M, 256, 0, stream>>>(
        x, b_in, b_rec, W_gate, b_gate, W_fc, b_fc, wtab, (float*)d_out);
}
